// BasinCoupledAttention_53214644798127
// MI455X (gfx1250) — matrix operand order repaired and checked by reading, not compiled, not run on hardware
//
#include <hip/hip_runtime.h>

#define D_MODEL 1024
#define N_HEADS 16
#define HEAD_DIM 64
#define BATCH 2
#define SEQ 2048
#define NROWS (BATCH * SEQ)  // 4096

typedef __attribute__((ext_vector_type(4)))  float          v4f;
typedef __attribute__((ext_vector_type(8)))  float          v8f;
typedef __attribute__((ext_vector_type(4)))  int            v4i;
typedef __attribute__((ext_vector_type(4)))  unsigned short v4u;
typedef __attribute__((ext_vector_type(8)))  unsigned short v8u;
typedef __attribute__((ext_vector_type(16))) unsigned short v16u;
typedef __attribute__((ext_vector_type(16))) __bf16         v16bf;

// ---- CDNA5 async global->LDS staging (ASYNCcnt-tracked) ------------------

#if defined(__gfx1250__) && __has_builtin(__builtin_amdgcn_global_load_async_to_lds_b128)
#define HAS_ASYNC 1
#else
#define HAS_ASYNC 0
#endif

typedef __attribute__((address_space(1))) void as1v;
typedef __attribute__((address_space(3))) void as3v;
typedef __attribute__((address_space(1))) v4i  as1_v4i;
typedef __attribute__((address_space(3))) v4i  as3_v4i;

__device__ __forceinline__ void cp_b128(const unsigned short* g,
                                        unsigned short* l) {
#if HAS_ASYNC
  __builtin_amdgcn_global_load_async_to_lds_b128(
      (as1_v4i*)(as1v*)g, (as3_v4i*)(as3v*)l, 0, 0);
#else
  *(v8u*)l = *(const v8u*)g;
#endif
}

#if HAS_ASYNC
#if __has_builtin(__builtin_amdgcn_s_wait_asynccnt)
#define WAIT_ASYNC(n) __builtin_amdgcn_s_wait_asynccnt(n)
#else
#define WAIT_ASYNC(n) asm volatile("s_wait_asynccnt %0" ::"i"(n) : "memory")
#endif
#else
#define WAIT_ASYNC(n) ((void)0)
#endif

// ---- helpers -------------------------------------------------------------

__device__ __forceinline__ unsigned short f2bf(float f) {
  unsigned u = __builtin_bit_cast(unsigned, f);
  u += 0x7FFFu + ((u >> 16) & 1u);  // round-to-nearest-even
  return (unsigned short)(u >> 16);
}

__device__ __forceinline__ v16u join16(v8u lo, v8u hi) {
  return __builtin_shufflevector(lo, hi, 0, 1, 2, 3, 4, 5, 6, 7, 8, 9, 10, 11,
                                 12, 13, 14, 15);
}

__device__ __forceinline__ v8f wmma_bf16(v16u a, v16u b, v8f c) {
  return __builtin_amdgcn_wmma_f32_16x16x32_bf16(
      false, __builtin_bit_cast(v16bf, a), false, __builtin_bit_cast(v16bf, b),
      (short)0, c, false, false);
}

// A operand (16x32): lane holds row M=lane%16; halves K = hi*8+[0..7], 16+hi*8+[0..7]
__device__ __forceinline__ v16u frag_a(const unsigned short* rowp, int laneHi) {
  v8u lo = *(const v8u*)(rowp + laneHi * 8);
  v8u hi = *(const v8u*)(rowp + 16 + laneHi * 8);
  return join16(lo, hi);
}
// B operand (32x16 stored [n][k]): lane holds col N=lane%16; halves K = hi*8+[0..7], 16+hi*8+[0..7]
__device__ __forceinline__ v16u frag_b(const unsigned short* rowp, int laneHi) {
  v8u lo = *(const v8u*)(rowp + laneHi * 8);
  v8u hi = *(const v8u*)(rowp + 16 + laneHi * 8);
  return join16(lo, hi);
}

// ---- gate: sigmoid(basin @ Wb + bb) -------------------------------------

__global__ void k_gate(const float* __restrict__ basin,
                       const float* __restrict__ Wb,
                       const float* __restrict__ bb, float* __restrict__ gate) {
  int h = threadIdx.x;
  if (h < N_HEADS) {
    float t = bb[h];
    for (int d = 0; d < 64; ++d) t += basin[d] * Wb[d * N_HEADS + h];
    gate[h] = 1.f / (1.f + __expf(-t));
  }
}

// ---- one-time f32 -> bf16 conversion (x) --------------------------------

__global__ __launch_bounds__(256) void k_cvt(const float* __restrict__ src,
                                             unsigned short* __restrict__ dst,
                                             int n4) {
  int i = blockIdx.x * 256 + threadIdx.x;
  if (i < n4) {
    v4f f = *(const v4f*)(src + (size_t)i * 4);
    v4u o;
    o.x = f2bf(f.x); o.y = f2bf(f.y); o.z = f2bf(f.z); o.w = f2bf(f.w);
    *(v4u*)(dst + (size_t)i * 4) = o;
  }
}

// ---- one-time weight f32[k][n] -> bf16 transposed [n][k] -----------------

__global__ __launch_bounds__(256) void k_cvt_wt(const float* __restrict__ W,
                                                unsigned short* __restrict__ Wt) {
  int idx = blockIdx.x * 256 + threadIdx.x;  // 1024 * 256 total
  int k = idx >> 8;
  int n = (idx & 255) * 4;
  v4f f = *(const v4f*)(W + (size_t)k * D_MODEL + n);
  Wt[(size_t)(n + 0) * D_MODEL + k] = f2bf(f.x);
  Wt[(size_t)(n + 1) * D_MODEL + k] = f2bf(f.y);
  Wt[(size_t)(n + 2) * D_MODEL + k] = f2bf(f.z);
  Wt[(size_t)(n + 3) * D_MODEL + k] = f2bf(f.w);
}

// ---- GEMM core: 8 waves, 128x128x32 tiles, async double-buffered LDS -----
// A: bf16 [M][K] row-major.  Wt: bf16 [N][K] (pre-transposed).
// Each thread stages 4 x b128 per tile (2 A + 2 B); per-wave tile 32x64.

#define GEMM_PROLOG()                                                        \
  __shared__ unsigned short As[2][128][40];                                  \
  __shared__ unsigned short Bs[2][128][40];                                  \
  const int tid = threadIdx.x;                                               \
  const int lane = tid & 31;                                                 \
  const int wave = tid >> 5;                                                 \
  const int laneHi = lane >> 4;                                              \
  const int lane16 = lane & 15;                                              \
  const int wr = wave >> 1, wc = wave & 1;                                   \
  const int m0 = blockIdx.y * 128;                                           \
  const int n0 = blockIdx.x * 128;                                           \
  const int sr = tid >> 2;           /* staging rows sr, sr+64 */            \
  const int scc = (tid & 3) * 8;     /* staging col (halves)   */            \
  v8f acc[2][4] = {};

#define GEMM_STAGE(buf, k0, Aglob, Bglob)                                    \
  {                                                                          \
    _Pragma("unroll") for (int i = 0; i < 2; ++i) {                          \
      int r = sr + i * 64;                                                   \
      cp_b128(Aglob + (size_t)(m0 + r) * D_MODEL + (k0) + scc,               \
              &As[buf][r][scc]);                                             \
      cp_b128(Bglob + (size_t)(n0 + r) * D_MODEL + (k0) + scc,               \
              &Bs[buf][r][scc]);                                             \
    }                                                                        \
  }

#define GEMM_MAINLOOP(Aglob, Bglob)                                          \
  GEMM_STAGE(0, 0, Aglob, Bglob);                                            \
  _Pragma("unroll 1") for (int k0 = 0; k0 < D_MODEL; k0 += 32) {             \
    const int buf = (k0 >> 5) & 1;                                           \
    if (k0 + 32 < D_MODEL) {                                                 \
      GEMM_STAGE(buf ^ 1, k0 + 32, Aglob, Bglob);                            \
      WAIT_ASYNC(4); /* in-order: tile k0's 4 copies done */                 \
    } else {                                                                 \
      WAIT_ASYNC(0);                                                         \
    }                                                                        \
    __syncthreads();                                                         \
    v16u a[2], bfr[4];                                                       \
    _Pragma("unroll") for (int i = 0; i < 2; ++i)                            \
        a[i] = frag_a(&As[buf][wr * 32 + i * 16 + lane16][0], laneHi);       \
    _Pragma("unroll") for (int j = 0; j < 4; ++j)                            \
        bfr[j] = frag_b(&Bs[buf][wc * 64 + j * 16 + lane16][0], laneHi);     \
    _Pragma("unroll") for (int i = 0; i < 2; ++i)                            \
        _Pragma("unroll") for (int j = 0; j < 4; ++j)                        \
            acc[i][j] = wmma_bf16(a[i], bfr[j], acc[i][j]);                  \
    __syncthreads();                                                         \
  }

// QKV projection: out = bf16 [B,H,S,Dh], weight selected by blockIdx.z
__global__ __launch_bounds__(256) void k_gemm_qkv(
    const unsigned short* __restrict__ Xb, const unsigned short* __restrict__ Wqt,
    const unsigned short* __restrict__ Wkt, const unsigned short* __restrict__ Wvt,
    const float* __restrict__ bq, const float* __restrict__ bk,
    const float* __restrict__ bv, unsigned short* __restrict__ Qh,
    unsigned short* __restrict__ Kh, unsigned short* __restrict__ Vh) {
  const int which = blockIdx.z;
  const unsigned short* Wt = (which == 0) ? Wqt : (which == 1) ? Wkt : Wvt;
  const float* bias = (which == 0) ? bq : (which == 1) ? bk : bv;
  unsigned short* Out = (which == 0) ? Qh : (which == 1) ? Kh : Vh;

  GEMM_PROLOG();
  GEMM_MAINLOOP(Xb, Wt);

#pragma unroll
  for (int i = 0; i < 2; ++i)
#pragma unroll
    for (int j = 0; j < 4; ++j) {
      int n = n0 + wc * 64 + j * 16 + lane16;
      int h = n >> 6, d = n & 63;
      float bb_ = bias[n];
#pragma unroll
      for (int g = 0; g < 8; ++g) {
        int m = m0 + wr * 32 + i * 16 + g + laneHi * 8;
        int bi = m >> 11, s = m & 2047;
        Out[(((size_t)(bi * N_HEADS + h) * SEQ) + s) * HEAD_DIM + d] =
            f2bf(acc[i][j][g] + bb_);
      }
    }
}

// Output projection: Y(f32) = AO(bf16) @ Wo + bo
__global__ __launch_bounds__(256) void k_gemm_out(
    const unsigned short* __restrict__ AO, const unsigned short* __restrict__ Wot,
    const float* __restrict__ bo, float* __restrict__ Y) {
  GEMM_PROLOG();
  GEMM_MAINLOOP(AO, Wot);

#pragma unroll
  for (int i = 0; i < 2; ++i)
#pragma unroll
    for (int j = 0; j < 4; ++j) {
      int n = n0 + wc * 64 + j * 16 + lane16;
      float bb_ = bo[n];
#pragma unroll
      for (int g = 0; g < 8; ++g) {
        int m = m0 + wr * 32 + i * 16 + g + laneHi * 8;
        Y[(size_t)m * D_MODEL + n] = acc[i][j][g] + bb_;
      }
    }
}

// ---- flash attention over 64-row q tiles ---------------------------------

__global__ __launch_bounds__(128) void k_attn(
    const unsigned short* __restrict__ Qh, const unsigned short* __restrict__ Kh,
    const unsigned short* __restrict__ Vh, const float* __restrict__ gate,
    unsigned short* __restrict__ AO) {
  __shared__ unsigned short Qs[64][72];  // [q][d]
  __shared__ unsigned short Ks[64][72];  // [krow][d]   (B layout for Q.K^T)
  __shared__ unsigned short Vt[64][72];  // [d][krow]   (B layout for P.V)
  __shared__ unsigned short Ps[64][72];  // [q][k], wave-private row stripes

  const int tid = threadIdx.x;
  const int lane = tid & 31;
  const int wave = tid >> 5;
  const int laneHi = lane >> 4;
  const int lane16 = lane & 15;

  const int qt = blockIdx.x;          // q tile (64 rows)
  const int bh = blockIdx.y;          // b*16 + h
  const int b = bh >> 4, h = bh & 15;

  const unsigned short* Qb = Qh + (size_t)bh * SEQ * HEAD_DIM;
  const unsigned short* Kb = Kh + (size_t)bh * SEQ * HEAD_DIM;
  const unsigned short* Vb = Vh + (size_t)bh * SEQ * HEAD_DIM;

#pragma unroll
  for (int i = 0; i < 4; ++i) {  // Q tile: 512 x b128
    int idx = tid + i * 128;
    int r = idx >> 3, c = (idx & 7) * 8;
    cp_b128(Qb + (size_t)(qt * 64 + r) * HEAD_DIM + c, &Qs[r][c]);
  }
  WAIT_ASYNC(0);
  __syncthreads();

  v16u qa[2];
#pragma unroll
  for (int cc = 0; cc < 2; ++cc)
    qa[cc] = frag_a(&Qs[wave * 16 + lane16][cc * 32], laneHi);

  const float scale = gate[h] * 0.125f;  // gate / sqrt(64)
  float m_i[8], l_i[8];
#pragma unroll
  for (int g = 0; g < 8; ++g) { m_i[g] = -1e30f; l_i[g] = 0.f; }
  v8f o[4] = {};

  for (int kt = 0; kt <= qt; ++kt) {
#pragma unroll
    for (int i = 0; i < 4; ++i) {  // K tile: async b128 copies
      int idx = tid + i * 128;
      int r = idx >> 3, c = (idx & 7) * 8;
      cp_b128(Kb + (size_t)(kt * 64 + r) * HEAD_DIM + c, &Ks[r][c]);
      if (i == 0) {  // speculative prefetch of next k-tile (stays in ws)
        __builtin_prefetch(Kb + (size_t)((kt + 1) * 64 + r) * HEAD_DIM + c, 0, 1);
        __builtin_prefetch(Vb + (size_t)((kt + 1) * 64 + r) * HEAD_DIM + c, 0, 1);
      }
    }
#pragma unroll
    for (int i = 0; i < 8; ++i) {  // V tile: transpose-scatter into [d][krow]
      int idx = tid + i * 128;
      int r = idx >> 4, c = (idx & 15) * 4;
      v4u vv = *(const v4u*)(Vb + (size_t)(kt * 64 + r) * HEAD_DIM + c);
      Vt[c + 0][r] = vv.x; Vt[c + 1][r] = vv.y;
      Vt[c + 2][r] = vv.z; Vt[c + 3][r] = vv.w;
    }
    WAIT_ASYNC(0);
    __syncthreads();

    // scores: 16 q-rows x 64 k-cols per wave
    v8f sc[4] = {};
#pragma unroll
    for (int j = 0; j < 4; ++j) {
      const unsigned short* krow = &Ks[j * 16 + lane16][0];
#pragma unroll
      for (int cc = 0; cc < 2; ++cc)
        sc[j] = wmma_bf16(qa[cc], frag_b(krow + cc * 32, laneHi), sc[j]);
    }
#pragma unroll
    for (int j = 0; j < 4; ++j)
#pragma unroll
      for (int g = 0; g < 8; ++g) sc[j][g] = sc[j][g] * scale;

    if (kt == qt) {  // causal mask only touches the diagonal tile
#pragma unroll
      for (int j = 0; j < 4; ++j)
#pragma unroll
        for (int g = 0; g < 8; ++g) {
          int ql = wave * 16 + g + laneHi * 8;
          int kl = j * 16 + lane16;
          if (kl > ql) sc[j][g] = -1e30f;
        }
    }

    // online softmax; C-frag row = g + 8*laneHi, cols striped over 16 lanes
#pragma unroll
    for (int g = 0; g < 8; ++g) {
      float rmax = fmaxf(fmaxf(sc[0][g], sc[1][g]), fmaxf(sc[2][g], sc[3][g]));
      rmax = fmaxf(rmax, __shfl_xor(rmax, 1));
      rmax = fmaxf(rmax, __shfl_xor(rmax, 2));
      rmax = fmaxf(rmax, __shfl_xor(rmax, 4));
      rmax = fmaxf(rmax, __shfl_xor(rmax, 8));
      float mnew = fmaxf(m_i[g], rmax);
      float corr = __expf(m_i[g] - mnew);
      float rsum = 0.f;
#pragma unroll
      for (int j = 0; j < 4; ++j) {
        float p = __expf(sc[j][g] - mnew);
        sc[j][g] = p;
        rsum += p;
      }
      rsum += __shfl_xor(rsum, 1);
      rsum += __shfl_xor(rsum, 2);
      rsum += __shfl_xor(rsum, 4);
      rsum += __shfl_xor(rsum, 8);
      l_i[g] = l_i[g] * corr + rsum;
      m_i[g] = mnew;
#pragma unroll
      for (int dj = 0; dj < 4; ++dj) o[dj][g] = o[dj][g] * corr;
    }

    // P (C-layout regs) -> wave-private LDS stripe -> A-layout frags
#pragma unroll
    for (int j = 0; j < 4; ++j)
#pragma unroll
      for (int g = 0; g < 8; ++g)
        Ps[wave * 16 + g + laneHi * 8][j * 16 + lane16] = f2bf(sc[j][g]);

#pragma unroll
    for (int cc = 0; cc < 2; ++cc) {
      v16u pa = frag_a(&Ps[wave * 16 + lane16][cc * 32], laneHi);
#pragma unroll
      for (int dj = 0; dj < 4; ++dj) {
        const unsigned short* vrow = &Vt[dj * 16 + lane16][0];
        o[dj] = wmma_bf16(pa, frag_b(vrow + cc * 32, laneHi), o[dj]);
      }
    }
    __syncthreads();
  }

  // normalize + write bf16 attn-out in [B*S, D_MODEL] (concat heads)
#pragma unroll
  for (int dj = 0; dj < 4; ++dj) {
    int d = dj * 16 + lane16;
#pragma unroll
    for (int g = 0; g < 8; ++g) {
      int s = qt * 64 + wave * 16 + g + laneHi * 8;
      AO[((size_t)(b * SEQ + s)) * D_MODEL + h * HEAD_DIM + d] =
          f2bf(o[dj][g] / l_i[g]);
    }
  }
}

// ---- launch --------------------------------------------------------------

extern "C" void kernel_launch(void* const* d_in, const int* in_sizes, int n_in,
                              void* d_out, int out_size, void* d_ws,
                              size_t ws_size, hipStream_t stream) {
  (void)in_sizes; (void)n_in; (void)out_size; (void)ws_size;
  const float* x     = (const float*)d_in[0];
  const float* basin = (const float*)d_in[1];
  // d_in[2] = causal mask (bool) — structure known, not read
  const float* Wq = (const float*)d_in[3];
  const float* bq = (const float*)d_in[4];
  const float* Wk = (const float*)d_in[5];
  const float* bk = (const float*)d_in[6];
  const float* Wv = (const float*)d_in[7];
  const float* bv = (const float*)d_in[8];
  const float* Wo = (const float*)d_in[9];
  const float* bo = (const float*)d_in[10];
  const float* Wb = (const float*)d_in[11];
  const float* bb = (const float*)d_in[12];
  float* out = (float*)d_out;

  char* ws = (char*)d_ws;
  const size_t MAT = (size_t)NROWS * D_MODEL * sizeof(unsigned short);    // 8 MiB
  const size_t WMAT = (size_t)D_MODEL * D_MODEL * sizeof(unsigned short); // 2 MiB
  unsigned short* Xb  = (unsigned short*)(ws + 0 * MAT);
  unsigned short* Qh  = (unsigned short*)(ws + 1 * MAT);
  unsigned short* Kh  = (unsigned short*)(ws + 2 * MAT);
  unsigned short* Vh  = (unsigned short*)(ws + 3 * MAT);
  unsigned short* AO  = (unsigned short*)(ws + 4 * MAT);
  unsigned short* Wqt = (unsigned short*)(ws + 5 * MAT);
  unsigned short* Wkt = (unsigned short*)(ws + 5 * MAT + 1 * WMAT);
  unsigned short* Wvt = (unsigned short*)(ws + 5 * MAT + 2 * WMAT);
  unsigned short* Wot = (unsigned short*)(ws + 5 * MAT + 3 * WMAT);
  float* gate = (float*)(ws + 5 * MAT + 4 * WMAT);

  k_gate<<<dim3(1), dim3(32), 0, stream>>>(basin, Wb, bb, gate);

  // one-time conversions: x -> bf16; weights -> bf16 transposed [n][k]
  k_cvt<<<dim3((NROWS * D_MODEL / 4) / 256), dim3(256), 0, stream>>>(
      x, Xb, NROWS * D_MODEL / 4);
  k_cvt_wt<<<dim3((D_MODEL * D_MODEL / 4) / 256), dim3(256), 0, stream>>>(Wq, Wqt);
  k_cvt_wt<<<dim3((D_MODEL * D_MODEL / 4) / 256), dim3(256), 0, stream>>>(Wk, Wkt);
  k_cvt_wt<<<dim3((D_MODEL * D_MODEL / 4) / 256), dim3(256), 0, stream>>>(Wv, Wvt);
  k_cvt_wt<<<dim3((D_MODEL * D_MODEL / 4) / 256), dim3(256), 0, stream>>>(Wo, Wot);

  k_gemm_qkv<<<dim3(D_MODEL / 128, NROWS / 128, 3), dim3(256), 0, stream>>>(
      Xb, Wqt, Wkt, Wvt, bq, bk, bv, Qh, Kh, Vh);
  k_attn<<<dim3(SEQ / 64, BATCH * N_HEADS), dim3(128), 0, stream>>>(
      Qh, Kh, Vh, gate, AO);
  k_gemm_out<<<dim3(D_MODEL / 128, NROWS / 128), dim3(256), 0, stream>>>(
      AO, Wot, bo, out);
}
